// AttentiveFP_68487548502615
// MI455X (gfx1250) — matrix operand order repaired and checked by reading, not compiled, not run on hardware
//
#include <hip/hip_runtime.h>
#include <math.h>

typedef __attribute__((ext_vector_type(16))) __bf16 v16bf;
typedef __attribute__((ext_vector_type(8)))  float  v8f;
typedef __attribute__((ext_vector_type(4)))  unsigned int v4u;
typedef __attribute__((ext_vector_type(8)))  int v8i_;
typedef __attribute__((ext_vector_type(4)))  int v4i_;

// ---------------------------------------------------------------------------
// TDM: stage a [128 rows x 32 cols] fp32 tile of A[M,K] (row-major) into LDS
// at byte offset lds_off, with 4-DWORD padding every 32 DWORDs (row pitch = 36
// floats) so fragment reads are bank-spread. OOB rows/cols return zeros.
// D# layout per CDNA5 ISA 8.3/8.4. Issued per-wave (gate to wave 0).
// This toolchain exposes the 6-arg builtin:
//   (uint32x4 g0, int32x8 g1, int32x4 g2, int32x4 g3, int32x8 pad, i32 cpol)
// ---------------------------------------------------------------------------
__device__ __forceinline__ void tdm_load_a_tile(const float* A, int M, int K,
                                                int m0, int k0, unsigned lds_off) {
  unsigned long long ga = (unsigned long long)(const void*)(A + (size_t)m0 * K + k0);
  v4u g0;
  g0[0] = 1u;                                       // count=1 (valid user D#)
  g0[1] = lds_off;                                  // lds_addr [63:32]
  g0[2] = (unsigned)(ga & 0xffffffffu);             // global_addr [95:64]
  g0[3] = (unsigned)((ga >> 32) & 0x1ffffffu)       // global_addr [120:96]
        | (2u << 30);                               // type=2 ("image") [127:126]
  unsigned dim0 = (unsigned)(K - k0);               // remaining row elems (OOB->0)
  unsigned dim1 = (unsigned)(M - m0);               // remaining rows     (OOB->0)
  unsigned long long s0 = (unsigned long long)K;    // tensor_dim0_stride
  v8i_ g1;
  g1[0] = (int)((2u << 16)      // data_size = 4B
              | (1u << 20)      // pad_enable
              | (4u << 22)      // pad_interval code 4 = 32 DWORDs
              | (3u << 25));    // pad_amount  code 3 = 4 DWORDs -> pitch 36 f32
  g1[1] = (int)((dim0 & 0xffffu) << 16);                   // tensor_dim0[15:0]
  g1[2] = (int)((dim0 >> 16) | ((dim1 & 0xffffu) << 16));  // dim0[31:16]|dim1[15:0]
  g1[3] = (int)((dim1 >> 16) | (32u << 16));               // dim1[31:16]|tile_dim0=32
  g1[4] = (int)128;                                        // tile_dim1=128, tile_dim2=0
  g1[5] = (int)(s0 & 0xffffffffu);                         // dim0_stride[31:0]
  g1[6] = (int)((s0 >> 32) & 0xffffu);                     // dim0_stride[47:32]
  g1[7] = 0;
  v4i_ z4 = {0, 0, 0, 0};
  v8i_ z8 = {0, 0, 0, 0, 0, 0, 0, 0};
  __builtin_amdgcn_tensor_load_to_lds(g0, g1, z4, z4, z8, 0);
}

// ---------------------------------------------------------------------------
// Tiled bf16 WMMA GEMM: C[M,Nc] = A[M,K] @ B[K,Nc]  (A,B,C fp32 row-major)
// Block = 256 threads (8 waves). BM=128 (wave w owns rows w*16..), BN=64
// (4 WMMA col-tiles/wave), BK=32. A staged fp32 via TDM (double-buffered,
// s_wait_tensorcnt), B staged transposed bf16 so both fragment reads are
// contiguous ds_load_b128. Fragment layouts used here:
//   A 16x32: lane L -> row L&15, kb=(L>>4)*16; elem e -> K=kb+e
//   B 32x16: lane L -> col L&15, kb=(L>>4)*16; elem i -> K=kb+i
//   C 16x16: lane L -> col L&15, vgpr r -> row r + 8*(L>>4)
// ---------------------------------------------------------------------------
#define GE_BM 128
#define GE_BN 64
#define GE_BK 32
#define GE_APITCH 36   // floats per LDS A row (32 + TDM pad 4)

__global__ __launch_bounds__(256) void gemm_bf16(const float* __restrict__ A,
                                                 const float* __restrict__ B,
                                                 float* __restrict__ C,
                                                 int M, int K, int Nc) {
  __shared__ float  AsF[2][GE_BM * GE_APITCH];   // 36 KB (TDM destination)
  __shared__ __bf16 BsT[GE_BN][GE_BK + 8];       // 5 KB, transposed B tile
  const int tid  = threadIdx.x;
  const int lane = tid & 31;
  const int wv   = tid >> 5;
  const int m0   = blockIdx.x * GE_BM;
  const int n0   = blockIdx.y * GE_BN;
  const int mrow = wv * 16;
  const int nk   = K / GE_BK;

  v8f acc[GE_BN / 16] = {};

  if (wv == 0)   // prime buffer 0
    tdm_load_a_tile(A, M, K, m0, 0, (unsigned)(size_t)&AsF[0][0]);

  for (int ks = 0; ks < nk; ++ks) {
    const int k0 = ks * GE_BK;
    __syncthreads();   // BsT from previous iter fully consumed
    // B tile: coalesced global read, transposed LDS write
    #pragma unroll 4
    for (int t = tid; t < GE_BK * GE_BN; t += 256) {
      int r = t / GE_BN, c = t % GE_BN;   // r = k, c = n
      BsT[c][r] = (__bf16)B[(size_t)(k0 + r) * Nc + (n0 + c)];
    }
    if (wv == 0) {
      if (ks + 1 < nk)
        tdm_load_a_tile(A, M, K, m0, k0 + GE_BK,
                        (unsigned)(size_t)&AsF[(ks + 1) & 1][0]);
      // A(ks) resident when <=1 newer TDM op outstanding (in-order completion)
      if (ks + 1 < nk) __builtin_amdgcn_s_wait_tensorcnt((short)1);
      else             __builtin_amdgcn_s_wait_tensorcnt((short)0);
    }
    __syncthreads();   // BsT ready for all; A(ks) ready (wave0 gated the wait)

    const float* Ar = &AsF[ks & 1][(mrow + (lane & 15)) * GE_APITCH];
    const int kbA = (lane >> 4) << 4;
    v16bf af;
    #pragma unroll
    for (int e = 0; e < 8; ++e) {
      af[e]     = (__bf16)Ar[kbA + e];
      af[8 + e] = (__bf16)Ar[kbA + 8 + e];
    }
    #pragma unroll
    for (int nt = 0; nt < GE_BN / 16; ++nt) {
      const __bf16* Br = &BsT[nt * 16 + (lane & 15)][(lane >> 4) << 4];
      v16bf bfrag;
      #pragma unroll
      for (int i = 0; i < 16; ++i) bfrag[i] = Br[i];
      acc[nt] = __builtin_amdgcn_wmma_f32_16x16x32_bf16(
          false, af, false, bfrag, (short)0, acc[nt], false, false);
    }
  }

  const int col16 = lane & 15;
  const int rbase = (lane >> 4) << 3;
  #pragma unroll
  for (int nt = 0; nt < GE_BN / 16; ++nt) {
    #pragma unroll
    for (int r = 0; r < 8; ++r) {
      int grow = m0 + mrow + rbase + r;
      if (grow < M) C[(size_t)grow * Nc + n0 + nt * 16 + col16] = acc[nt][r];
    }
  }
}

// ---------------------------------------------------------------------------
// Per-layer init: out[i,c] = bias[c]; segment-max keys = 0 (== -inf); denom = 0
// ---------------------------------------------------------------------------
__global__ void init_layer_kernel(float* __restrict__ out, const float* __restrict__ bias,
                                  unsigned* __restrict__ mkey, float* __restrict__ denom,
                                  int Nn, int Hh) {
  size_t t = (size_t)blockIdx.x * blockDim.x + threadIdx.x;
  size_t total = (size_t)Nn * Hh;
  if (t < total) out[t] = bias[t % Hh];
  if (t < (size_t)Nn) { mkey[t] = 0u; denom[t] = 0.f; }
}

__global__ void relu_kernel(float* __restrict__ p, size_t n) {
  size_t t = (size_t)blockIdx.x * blockDim.x + threadIdx.x;
  if (t < n) { float v = p[t]; p[t] = v > 0.f ? v : 0.f; }
}

__global__ void zero_kernel(float* __restrict__ p, size_t n) {
  size_t t = (size_t)blockIdx.x * blockDim.x + threadIdx.x;
  if (t < n) p[t] = 0.f;
}

// ---------------------------------------------------------------------------
// Edge scores: one wave per edge (self-loops appended virtually: e>=E -> (i,i)).
// e = leaky_relu(xp[src].a_src + xp[dst].a_dst); segment-max via order-preserving
// uint keys + atomicMax (monotone map: f>=0 -> bits|0x80000000, f<0 -> ~bits).
// ---------------------------------------------------------------------------
__global__ __launch_bounds__(256) void edge_score_kernel(
    const float* __restrict__ xp, const int* __restrict__ esrc, const int* __restrict__ edst,
    const float* __restrict__ a_src, const float* __restrict__ a_dst,
    float* __restrict__ escore, unsigned* __restrict__ mkey,
    int Etot, int Eorig, int Hh) {
  int wid = blockIdx.x * 8 + (threadIdx.x >> 5);
  if (wid >= Etot) return;
  int lane = threadIdx.x & 31;
  int s, d;
  if (wid < Eorig) { s = esrc[wid]; d = edst[wid]; }
  else             { s = wid - Eorig; d = s; }
  const float* xs = xp + (size_t)s * Hh;
  const float* xd = xp + (size_t)d * Hh;
  float acc = 0.f;
  for (int c = lane; c < Hh; c += 32)
    acc = fmaf(xs[c], a_src[c], fmaf(xd[c], a_dst[c], acc));
  #pragma unroll
  for (int off = 16; off > 0; off >>= 1) acc += __shfl_xor(acc, off, 32);
  if (lane == 0) {
    float v = acc > 0.f ? acc : 0.2f * acc;   // LeakyReLU(0.2)
    escore[wid] = v;
    unsigned b   = __float_as_uint(v);
    unsigned key = (b & 0x80000000u) ? ~b : (b | 0x80000000u);
    atomicMax(&mkey[d], key);
  }
}

// p = exp(e - m[dst]); denom[dst] += p  (in-place over escore)
__global__ void edge_weight_kernel(float* __restrict__ escore,
                                   const unsigned* __restrict__ mkey,
                                   float* __restrict__ denom,
                                   const int* __restrict__ edst, int Etot, int Eorig) {
  int e = blockIdx.x * blockDim.x + threadIdx.x;
  if (e >= Etot) return;
  int d = (e < Eorig) ? edst[e] : (e - Eorig);
  unsigned key = mkey[d];
  unsigned b   = (key & 0x80000000u) ? (key & 0x7fffffffu) : ~key;
  float m = __uint_as_float(b);
  float p = __expf(escore[e] - m);
  escore[e] = p;
  unsafeAtomicAdd(&denom[d], p);
}

// out[dst] += (p/denom[dst]) * xp[src]   (one wave per edge, HW f32 atomics)
__global__ __launch_bounds__(256) void aggregate_kernel(
    const float* __restrict__ xp, const float* __restrict__ escore,
    const float* __restrict__ denom, const int* __restrict__ esrc,
    const int* __restrict__ edst, float* __restrict__ out,
    int Etot, int Eorig, int Hh) {
  int wid = blockIdx.x * 8 + (threadIdx.x >> 5);
  if (wid >= Etot) return;
  int lane = threadIdx.x & 31;
  int s, d;
  if (wid < Eorig) { s = esrc[wid]; d = edst[wid]; }
  else             { s = wid - Eorig; d = s; }
  float alpha = escore[wid] / denom[d];
  const float* xs = xp + (size_t)s * Hh;
  float* od = out + (size_t)d * Hh;
  for (int c = lane; c < Hh; c += 32) unsafeAtomicAdd(&od[c], alpha * xs[c]);
}

// ---------------------------------------------------------------------------
// Fused GRU (h0=0) + global_add_pool. Block handles a 16-row node tile:
// gi[16,768] = X_tile @ W_ih^T via 48 WMMA col-tiles (8 waves x 6), gates in
// LDS, h=(1-z)*tanh(...) scattered with f32 atomics into pooled[batch[i]].
// Assumes H == 256 (3H = 768).
// ---------------------------------------------------------------------------
__global__ __launch_bounds__(256) void gru_pool_kernel(
    const float* __restrict__ X, const float* __restrict__ W_ih,
    const float* __restrict__ b_ih, const float* __restrict__ b_hh,
    const int* __restrict__ batch, float* __restrict__ pooled, int Nn) {
  __shared__ __bf16 As[16][40];
  __shared__ float  gi[16][776];
  const int tid  = threadIdx.x;
  const int lane = tid & 31;
  const int wv   = tid >> 5;
  const int m0   = blockIdx.x * 16;

  v8f acc[6] = {};
  for (int k0 = 0; k0 < 256; k0 += 32) {
    __syncthreads();
    for (int t = tid; t < 16 * 32; t += 256) {
      int r = t >> 5, c = t & 31;
      int gr = m0 + r;
      As[r][c] = (__bf16)((gr < Nn) ? X[(size_t)gr * 256 + k0 + c] : 0.f);
    }
    __syncthreads();
    v16bf af;
    {
      int row = lane & 15;
      int kb  = (lane >> 4) << 4;
      #pragma unroll
      for (int e = 0; e < 8; ++e) {
        af[e]     = As[row][kb + e];
        af[8 + e] = As[row][kb + 8 + e];
      }
    }
    #pragma unroll
    for (int j = 0; j < 6; ++j) {
      int nt  = wv * 6 + j;                 // gate-column tile in [0,48)
      int col = nt * 16 + (lane & 15);      // row of W_ih (gi = X @ W_ih^T)
      int kb  = k0 + ((lane >> 4) << 4);
      v16bf bfrag;
      #pragma unroll
      for (int i = 0; i < 16; ++i)
        bfrag[i] = (__bf16)W_ih[(size_t)col * 256 + kb + i];
      acc[j] = __builtin_amdgcn_wmma_f32_16x16x32_bf16(
          false, af, false, bfrag, (short)0, acc[j], false, false);
    }
  }

  const int col16 = lane & 15;
  const int rbase = (lane >> 4) << 3;
  #pragma unroll
  for (int j = 0; j < 6; ++j) {
    int nt = wv * 6 + j;
    #pragma unroll
    for (int r = 0; r < 8; ++r) gi[rbase + r][nt * 16 + col16] = acc[j][r];
  }
  __syncthreads();

  for (int t = tid; t < 16 * 256; t += 256) {
    int row = t >> 8, c = t & 255;
    int gr = m0 + row;
    if (gr >= Nn) continue;
    float grv = gi[row][c]       + b_ih[c]       + b_hh[c];
    float gzv = gi[row][256 + c] + b_ih[256 + c] + b_hh[256 + c];
    float gnv = gi[row][512 + c] + b_ih[512 + c];
    float rg = 1.f / (1.f + __expf(-grv));
    float zg = 1.f / (1.f + __expf(-gzv));
    float ng = tanhf(gnv + rg * b_hh[512 + c]);
    float hv = (1.f - zg) * ng;               // + z*h0 with h0=0
    unsafeAtomicAdd(&pooled[(size_t)batch[gr] * 256 + c], hv);
  }
}

// ---------------------------------------------------------------------------
// super = relu(mean_g(pooled) @ Wf^T + bf); out = concat([pooled, super], dim=1)
// Single block of 256 threads (H == 256).
// ---------------------------------------------------------------------------
__global__ __launch_bounds__(256) void final_kernel(
    const float* __restrict__ pooled, const float* __restrict__ Wf,
    const float* __restrict__ bfc, float* __restrict__ out, int Gg, int Hh) {
  __shared__ float smean[256];
  __shared__ float ssuper[256];
  int tid = threadIdx.x;
  float sum = 0.f;
  for (int g = 0; g < Gg; ++g) sum += pooled[(size_t)g * Hh + tid];
  smean[tid] = sum / (float)Gg;
  __syncthreads();
  float accd = bfc[tid];
  for (int k = 0; k < Hh; ++k) accd = fmaf(smean[k], Wf[(size_t)tid * Hh + k], accd);
  ssuper[tid] = accd > 0.f ? accd : 0.f;
  __syncthreads();
  int total = Gg * 2 * Hh;
  for (int t = tid; t < total; t += 256) {
    int g = t / (2 * Hh);
    int c = t % (2 * Hh);
    out[t] = (c < Hh) ? pooled[(size_t)g * Hh + c] : ssuper[c - Hh];
  }
}

// ---------------------------------------------------------------------------
extern "C" void kernel_launch(void* const* d_in, const int* in_sizes, int n_in,
                              void* d_out, int out_size, void* d_ws, size_t ws_size,
                              hipStream_t stream) {
  (void)n_in; (void)ws_size;
  const float* x      = (const float*)d_in[0];
  const int*   eidx   = (const int*)d_in[1];
  const int*   batch  = (const int*)d_in[2];
  const float* W0     = (const float*)d_in[3];
  const float* a_src0 = (const float*)d_in[4];
  const float* a_dst0 = (const float*)d_in[5];
  const float* b0     = (const float*)d_in[6];
  const float* W1     = (const float*)d_in[7];
  const float* a_src1 = (const float*)d_in[8];
  const float* a_dst1 = (const float*)d_in[9];
  const float* b1     = (const float*)d_in[10];
  const float* W_ih   = (const float*)d_in[11];
  /* d_in[12] = W_hh: unused, h0 == 0 -> hidden contribution is b_hh only */
  const float* b_ih   = (const float*)d_in[13];
  const float* b_hh   = (const float*)d_in[14];
  const float* Wf     = (const float*)d_in[15];
  const float* bfc    = (const float*)d_in[16];
  float* out = (float*)d_out;

  const int Nn   = in_sizes[2];
  const int E    = in_sizes[1] / 2;
  const int Hh   = in_sizes[4];
  const int IN_  = in_sizes[0] / Nn;
  const int Gg   = out_size / (2 * Hh);
  const int Etot = E + Nn;
  const int* esrc = eidx;
  const int* edst = eidx + E;

  // workspace carve-out (~106 MB)
  char* ws = (char*)d_ws;
  size_t nh_bytes = (size_t)Nn * Hh * sizeof(float);
  float*    bufA   = (float*)ws;    ws += nh_bytes;
  float*    bufB   = (float*)ws;    ws += nh_bytes;
  float*    escore = (float*)ws;    ws += (size_t)Etot * sizeof(float);
  unsigned* mkey   = (unsigned*)ws; ws += (size_t)Nn * sizeof(unsigned);
  float*    denom  = (float*)ws;    ws += (size_t)Nn * sizeof(float);
  float*    pooled = (float*)ws;    ws += (size_t)Gg * Hh * sizeof(float);

  dim3 blk(256);
  dim3 gemm_grid((Nn + GE_BM - 1) / GE_BM, Hh / GE_BN);
  int eblocks   = (Etot + 7) / 8;
  int nh_blocks = (int)(((size_t)Nn * Hh + 255) / 256);

  // ---- GAT layer 0: xp0 = x @ W0 ; edge softmax ; aggregate ; relu ----
  gemm_bf16<<<gemm_grid, blk, 0, stream>>>(x, W0, bufA, Nn, IN_, Hh);
  init_layer_kernel<<<nh_blocks, blk, 0, stream>>>(bufB, b0, mkey, denom, Nn, Hh);
  edge_score_kernel<<<eblocks, blk, 0, stream>>>(bufA, esrc, edst, a_src0, a_dst0,
                                                 escore, mkey, Etot, E, Hh);
  edge_weight_kernel<<<(Etot + 255) / 256, blk, 0, stream>>>(escore, mkey, denom,
                                                             edst, Etot, E);
  aggregate_kernel<<<eblocks, blk, 0, stream>>>(bufA, escore, denom, esrc, edst,
                                                bufB, Etot, E, Hh);
  relu_kernel<<<nh_blocks, blk, 0, stream>>>(bufB, (size_t)Nn * Hh);

  // ---- GAT layer 1: xp1 = h0 @ W1 ; edge softmax ; aggregate ; relu ----
  gemm_bf16<<<gemm_grid, blk, 0, stream>>>(bufB, W1, bufA, Nn, Hh, Hh);
  init_layer_kernel<<<nh_blocks, blk, 0, stream>>>(bufB, b1, mkey, denom, Nn, Hh);
  edge_score_kernel<<<eblocks, blk, 0, stream>>>(bufA, esrc, edst, a_src1, a_dst1,
                                                 escore, mkey, Etot, E, Hh);
  edge_weight_kernel<<<(Etot + 255) / 256, blk, 0, stream>>>(escore, mkey, denom,
                                                             edst, Etot, E);
  aggregate_kernel<<<eblocks, blk, 0, stream>>>(bufA, escore, denom, esrc, edst,
                                                bufB, Etot, E, Hh);
  relu_kernel<<<nh_blocks, blk, 0, stream>>>(bufB, (size_t)Nn * Hh);

  // ---- GRU step (h0=0) fused with global_add_pool ----
  zero_kernel<<<(Gg * Hh + 255) / 256, blk, 0, stream>>>(pooled, (size_t)Gg * Hh);
  gru_pool_kernel<<<(Nn + 15) / 16, blk, 0, stream>>>(bufB, W_ih, b_ih, b_hh,
                                                      batch, pooled, Nn);

  // ---- super node FC + concat output ----
  final_kernel<<<1, blk, 0, stream>>>(pooled, Wf, bfc, out, Gg, Hh);
}
